// MultiHeadSelfAttention_57535381897598
// MI455X (gfx1250) — matrix operand order repaired and checked by reading, not compiled, not run on hardware
//
#include <hip/hip_runtime.h>
#include <hip/hip_bf16.h>

// Problem constants
#define B_  4
#define S_  2048
#define D_  1024
#define H_  16
#define HD_ 64

typedef __bf16 bf16_t;
typedef __attribute__((ext_vector_type(16))) __bf16 v16bf;
typedef __attribute__((ext_vector_type(8)))  __bf16 v8bf;
typedef __attribute__((ext_vector_type(4)))  __bf16 v4bf;
typedef __attribute__((ext_vector_type(8)))  float  v8f;
typedef __attribute__((ext_vector_type(4)))  float  v4f;

__device__ __forceinline__ v8bf ld8bf(const bf16_t* p) {
    return *reinterpret_cast<const v8bf*>(p);
}
__device__ __forceinline__ v16bf cat16(v8bf lo, v8bf hi) {
    return __builtin_shufflevector(lo, hi, 0,1,2,3,4,5,6,7,8,9,10,11,12,13,14,15);
}
__device__ __forceinline__ v8f wmma_bf16(v16bf a, v16bf b, v8f c) {
    return __builtin_amdgcn_wmma_f32_16x16x32_bf16(
        /*neg_a=*/false, a, /*neg_b=*/false, b,
        /*c_mod=*/(short)0, c, /*reuse_a=*/false, /*reuse_b=*/false);
}

// ---------------------------------------------------------------------------
// Kernel 1: x (f32) -> bf16, elementwise, 4 per thread
// ---------------------------------------------------------------------------
__global__ __launch_bounds__(256) void cvt_x_kernel(const float* __restrict__ x,
                                                    bf16_t* __restrict__ xb) {
    size_t i = ((size_t)blockIdx.x * 256 + threadIdx.x) * 4;
    v4f v = *reinterpret_cast<const v4f*>(x + i);
    v4bf r;
    r[0] = (__bf16)v[0]; r[1] = (__bf16)v[1]; r[2] = (__bf16)v[2]; r[3] = (__bf16)v[3];
    *reinterpret_cast<v4bf*>(xb + i) = r;
}

// ---------------------------------------------------------------------------
// Kernel 2: W (D x D, f32, k-major rows) -> W^T (N x K) bf16 via LDS transpose
// ---------------------------------------------------------------------------
__global__ __launch_bounds__(256) void cvt_wt_kernel(const float* __restrict__ W0,
                                                     const float* __restrict__ W1,
                                                     const float* __restrict__ W2,
                                                     const float* __restrict__ W3,
                                                     bf16_t* __restrict__ dstbase) {
    __shared__ float tile[16][17];
    const float* W = (blockIdx.z == 0) ? W0 : (blockIdx.z == 1) ? W1
                   : (blockIdx.z == 2) ? W2 : W3;
    bf16_t* dst = dstbase + (size_t)blockIdx.z * D_ * D_;
    int n0 = blockIdx.x * 16, k0 = blockIdx.y * 16;
    int tx = threadIdx.x & 15, ty = threadIdx.x >> 4;
    tile[ty][tx] = W[(size_t)(k0 + ty) * D_ + n0 + tx];
    __syncthreads();
    dst[(size_t)(n0 + ty) * D_ + k0 + tx] = (bf16_t)tile[tx][ty];
}

// ---------------------------------------------------------------------------
// Kernel 3: GEMM  dst = xb[8192 x 1024] @ Wt^T + bias  (bf16 WMMA, f32 accum)
// transposeV!=0 => store V per-head transposed: Vt[(b*H+h)*HD + d][s]
// grid (512, 8), block 256 (8 waves, one 16x16 tile each)
// ---------------------------------------------------------------------------
__global__ __launch_bounds__(256) void qkv_gemm_kernel(const bf16_t* __restrict__ xb,
                                                       const bf16_t* __restrict__ Wt,
                                                       const float* __restrict__ bias,
                                                       bf16_t* __restrict__ dst,
                                                       int transposeV) {
    const int lane = threadIdx.x & 31;
    const int wave = threadIdx.x >> 5;
    const int n16  = lane & 15;
    const int hs   = lane >> 4;
    const int tileM = blockIdx.x;
    const int tileN = blockIdx.y * 8 + wave;

    const bf16_t* aRow = xb + (size_t)(tileM * 16 + n16) * D_;
    const bf16_t* bRow = Wt + (size_t)(tileN * 16 + n16) * D_;

    v8f acc = {};
    for (int k0 = 0; k0 < D_; k0 += 32) {
        v16bf a = cat16(ld8bf(aRow + k0 + hs * 8), ld8bf(aRow + k0 + 16 + hs * 8));
        v16bf b = cat16(ld8bf(bRow + k0 + hs * 8), ld8bf(bRow + k0 + 16 + hs * 8));
        acc = wmma_bf16(a, b, acc);
        __builtin_prefetch((const void*)(aRow + k0 + 256), 0, 1);
        __builtin_prefetch((const void*)(bRow + k0 + 256), 0, 1);
    }

    const int n = tileN * 16 + n16;
    const float bv = bias[n];
    if (!transposeV) {
#pragma unroll
        for (int e = 0; e < 8; ++e) {
            int m = tileM * 16 + hs * 8 + e;
            dst[(size_t)m * D_ + n] = (bf16_t)(acc[e] + bv);
        }
    } else {
        int h = n >> 6, d = n & 63;
#pragma unroll
        for (int e = 0; e < 8; ++e) {
            int m = tileM * 16 + hs * 8 + e;
            int bb = m >> 11, s = m & (S_ - 1);
            dst[(size_t)((bb * H_ + h) * HD_ + d) * S_ + s] = (bf16_t)(acc[e] + bv);
        }
    }
}

// ---------------------------------------------------------------------------
// Kernel 4 (FUSED): scores -> softmax -> weights write -> attn = P @ V
// One workgroup = one (b,h) x 16 query rows. 256 threads = 8 waves.
// Phase 1: each wave computes 16 score tiles (16x16) over full S, A-frag
//          (the Q block) loaded once and reused for all 16 WMMA pairs.
// Phase 2: softmax via __shfl_xor half-group reduction + LDS cross-wave.
// Phase 3: single f32 write of weights to d_out; normalized P staged to LDS
//          as bf16 (row stride 2056 keeps 16B-aligned conflict-light loads).
// Phase 4: attn = P(16x2048) @ V(2048x64), split-K across wave quads,
//          partial sums combined through LDS, bf16 result to attnb.
// grid (S/16=128, B*H=64), block 256
// ---------------------------------------------------------------------------
#define PSTRIDE 2056   // 2056*2 = 4112 bytes = 257*16 per row

__global__ __launch_bounds__(256) void fused_attn_kernel(const bf16_t* __restrict__ Qb,
                                                         const bf16_t* __restrict__ Kb,
                                                         const bf16_t* __restrict__ Vt,
                                                         float* __restrict__ w,
                                                         bf16_t* __restrict__ attnb) {
    __shared__ bf16_t P[16 * PSTRIDE];       // 64.25 KiB normalized weights
    __shared__ float red[16][8];
    __shared__ float rowfin[16];
    __shared__ float rowinv[16];
    __shared__ float part[4][256];

    const int lane = threadIdx.x & 31;
    const int wave = threadIdx.x >> 5;
    const int n16  = lane & 15;
    const int hs   = lane >> 4;
    const int tileM = blockIdx.x;
    const int bh = blockIdx.y, b = bh >> 4, h = bh & 15;

    // ---- Phase 1: scores = Q_blk @ K^T (A-frags loaded once) ----
    const bf16_t* aRow = Qb + (size_t)(b * S_ + tileM * 16 + n16) * D_ + h * HD_;
    v16bf a0 = cat16(ld8bf(aRow + hs * 8),      ld8bf(aRow + 16 + hs * 8));
    v16bf a1 = cat16(ld8bf(aRow + 32 + hs * 8), ld8bf(aRow + 48 + hs * 8));

    v8f acc[16];
#pragma unroll
    for (int t = 0; t < 16; ++t) {
        const bf16_t* bRow =
            Kb + (size_t)(b * S_ + (wave * 16 + t) * 16 + n16) * D_ + h * HD_;
        v16bf b0 = cat16(ld8bf(bRow + hs * 8),      ld8bf(bRow + 16 + hs * 8));
        v16bf b1 = cat16(ld8bf(bRow + 32 + hs * 8), ld8bf(bRow + 48 + hs * 8));
        v8f c = {};
        c = wmma_bf16(a0, b0, c);
        acc[t] = wmma_bf16(a1, b1, c);
    }

    // ---- Phase 2a: scale + per-lane row max, half-group shuffle reduce ----
    float m[8];
#pragma unroll
    for (int e = 0; e < 8; ++e) m[e] = -3.4e38f;
#pragma unroll
    for (int t = 0; t < 16; ++t)
#pragma unroll
        for (int e = 0; e < 8; ++e) {
            float v = acc[t][e] * 0.125f;          // 1/sqrt(HD)
            acc[t][e] = v;
            m[e] = fmaxf(m[e], v);
        }
#pragma unroll
    for (int e = 0; e < 8; ++e) {
        m[e] = fmaxf(m[e], __shfl_xor(m[e], 1, 32));
        m[e] = fmaxf(m[e], __shfl_xor(m[e], 2, 32));
        m[e] = fmaxf(m[e], __shfl_xor(m[e], 4, 32));
        m[e] = fmaxf(m[e], __shfl_xor(m[e], 8, 32));
    }
    if (n16 == 0) {
#pragma unroll
        for (int e = 0; e < 8; ++e) red[hs * 8 + e][wave] = m[e];
    }
    __syncthreads();
    if (threadIdx.x < 16) {
        float v = red[threadIdx.x][0];
#pragma unroll
        for (int wv = 1; wv < 8; ++wv) v = fmaxf(v, red[threadIdx.x][wv]);
        rowfin[threadIdx.x] = v;
    }
    __syncthreads();

    // ---- Phase 2b: exp + row sum ----
    float rm[8], s[8];
#pragma unroll
    for (int e = 0; e < 8; ++e) { rm[e] = rowfin[hs * 8 + e]; s[e] = 0.f; }
#pragma unroll
    for (int t = 0; t < 16; ++t)
#pragma unroll
        for (int e = 0; e < 8; ++e) {
            float p = __expf(acc[t][e] - rm[e]);
            acc[t][e] = p;
            s[e] += p;
        }
#pragma unroll
    for (int e = 0; e < 8; ++e) {
        s[e] += __shfl_xor(s[e], 1, 32);
        s[e] += __shfl_xor(s[e], 2, 32);
        s[e] += __shfl_xor(s[e], 4, 32);
        s[e] += __shfl_xor(s[e], 8, 32);
    }
    if (n16 == 0) {
#pragma unroll
        for (int e = 0; e < 8; ++e) red[hs * 8 + e][wave] = s[e];
    }
    __syncthreads();
    if (threadIdx.x < 16) {
        float v = red[threadIdx.x][0];
#pragma unroll
        for (int wv = 1; wv < 8; ++wv) v += red[threadIdx.x][wv];
        rowinv[threadIdx.x] = 1.0f / v;
    }
    __syncthreads();

    // ---- Phase 3: normalize, write weights (f32) once, stage bf16 P to LDS
    float inv[8];
#pragma unroll
    for (int e = 0; e < 8; ++e) inv[e] = rowinv[hs * 8 + e];

    float* wbase = w + (size_t)bh * S_ * S_ + (size_t)(tileM * 16) * S_;
#pragma unroll
    for (int t = 0; t < 16; ++t) {
        const int col = wave * 256 + t * 16 + n16;
#pragma unroll
        for (int e = 0; e < 8; ++e) {
            const int r = hs * 8 + e;
            const float p = acc[t][e] * inv[e];
            wbase[(size_t)r * S_ + col] = p;
            P[r * PSTRIDE + col] = (bf16_t)p;
        }
    }
    __syncthreads();

    // ---- Phase 4: attn = P @ V, split-K across wave quads ----
    const int dt = wave & 3;      // d-tile 0..3
    const int kh = wave >> 2;     // K half 0/1
    const bf16_t* bRow = Vt + (size_t)(bh * HD_ + dt * 16 + n16) * S_ + kh * 1024;
    const bf16_t* aP = P + n16 * PSTRIDE + kh * 1024;

    v8f o = {};
    for (int k0 = 0; k0 < 1024; k0 += 32) {
        v16bf a  = cat16(ld8bf(aP + k0 + hs * 8), ld8bf(aP + k0 + 16 + hs * 8));
        v16bf bb = cat16(ld8bf(bRow + k0 + hs * 8), ld8bf(bRow + k0 + 16 + hs * 8));
        o = wmma_bf16(a, bb, o);
        __builtin_prefetch((const void*)(bRow + k0 + 512), 0, 1);
    }
    if (kh == 1) {
#pragma unroll
        for (int e = 0; e < 8; ++e) part[dt][e * 32 + lane] = o[e];
    }
    __syncthreads();
    if (kh == 0) {
#pragma unroll
        for (int e = 0; e < 8; ++e) {
            o[e] += part[dt][e * 32 + lane];
            int q = tileM * 16 + hs * 8 + e;
            attnb[(size_t)(b * S_ + q) * D_ + h * HD_ + dt * 16 + n16] = (bf16_t)o[e];
        }
    }
}

// ---------------------------------------------------------------------------
// Kernel 5: out = attn @ Wo^T + bo  (f32 output), grid (512, 8), block 256
// ---------------------------------------------------------------------------
__global__ __launch_bounds__(256) void out_gemm_kernel(const bf16_t* __restrict__ attnb,
                                                       const bf16_t* __restrict__ WoT,
                                                       const float* __restrict__ bo,
                                                       float* __restrict__ out) {
    const int lane = threadIdx.x & 31;
    const int wave = threadIdx.x >> 5;
    const int n16  = lane & 15;
    const int hs   = lane >> 4;
    const int tileM = blockIdx.x;
    const int tileN = blockIdx.y * 8 + wave;

    const bf16_t* aRow = attnb + (size_t)(tileM * 16 + n16) * D_;
    const bf16_t* bRow = WoT   + (size_t)(tileN * 16 + n16) * D_;

    v8f acc = {};
    for (int k0 = 0; k0 < D_; k0 += 32) {
        v16bf a = cat16(ld8bf(aRow + k0 + hs * 8), ld8bf(aRow + k0 + 16 + hs * 8));
        v16bf b = cat16(ld8bf(bRow + k0 + hs * 8), ld8bf(bRow + k0 + 16 + hs * 8));
        acc = wmma_bf16(a, b, acc);
        __builtin_prefetch((const void*)(aRow + k0 + 256), 0, 1);
        __builtin_prefetch((const void*)(bRow + k0 + 256), 0, 1);
    }

    const int n = tileN * 16 + n16;
    const float bv = bo[n];
#pragma unroll
    for (int e = 0; e < 8; ++e) {
        int m = tileM * 16 + hs * 8 + e;
        out[(size_t)m * D_ + n] = acc[e] + bv;
    }
}

// ---------------------------------------------------------------------------
// Workspace layout (bytes): 88 MiB total
// ---------------------------------------------------------------------------
static const size_t OFF_XB = 0;                              // 16 MiB  x bf16
static const size_t OFF_WT = OFF_XB + (size_t)B_*S_*D_*2;    //  8 MiB  4x W^T bf16
static const size_t OFF_QB = OFF_WT + (size_t)4*D_*D_*2;     // 16 MiB  Q bf16
static const size_t OFF_KB = OFF_QB + (size_t)B_*S_*D_*2;    // 16 MiB  K bf16
static const size_t OFF_VT = OFF_KB + (size_t)B_*S_*D_*2;    // 16 MiB  V^T bf16
static const size_t OFF_AT = OFF_VT + (size_t)B_*S_*D_*2;    // 16 MiB  attn bf16

extern "C" void kernel_launch(void* const* d_in, const int* in_sizes, int n_in,
                              void* d_out, int out_size, void* d_ws, size_t ws_size,
                              hipStream_t stream) {
    (void)in_sizes; (void)n_in; (void)out_size; (void)ws_size;
    const float* x  = (const float*)d_in[0];
    const float* Wq = (const float*)d_in[1];
    const float* bq = (const float*)d_in[2];
    const float* Wk = (const float*)d_in[3];
    const float* bk = (const float*)d_in[4];
    const float* Wv = (const float*)d_in[5];
    const float* bv = (const float*)d_in[6];
    const float* Wo = (const float*)d_in[7];
    const float* bo = (const float*)d_in[8];

    float* out     = (float*)d_out;
    float* weights = out + (size_t)B_ * S_ * D_;   // tuple order: (out, weights)

    char* ws = (char*)d_ws;
    bf16_t* xb    = (bf16_t*)(ws + OFF_XB);
    bf16_t* wtb   = (bf16_t*)(ws + OFF_WT);
    bf16_t* WqT   = wtb;
    bf16_t* WkT   = wtb + (size_t)1 * D_ * D_;
    bf16_t* WvT   = wtb + (size_t)2 * D_ * D_;
    bf16_t* WoT   = wtb + (size_t)3 * D_ * D_;
    bf16_t* Qb    = (bf16_t*)(ws + OFF_QB);
    bf16_t* Kb    = (bf16_t*)(ws + OFF_KB);
    bf16_t* Vt    = (bf16_t*)(ws + OFF_VT);
    bf16_t* attnb = (bf16_t*)(ws + OFF_AT);

    // 1) precision conversions
    cvt_x_kernel<<<(B_ * S_ * D_) / (256 * 4), 256, 0, stream>>>(x, xb);
    cvt_wt_kernel<<<dim3(D_ / 16, D_ / 16, 4), 256, 0, stream>>>(Wq, Wk, Wv, Wo, wtb);

    // 2) Q/K/V projections (V stored transposed per head)
    qkv_gemm_kernel<<<dim3((B_ * S_) / 16, D_ / 128), 256, 0, stream>>>(xb, WqT, bq, Qb, 0);
    qkv_gemm_kernel<<<dim3((B_ * S_) / 16, D_ / 128), 256, 0, stream>>>(xb, WkT, bk, Kb, 0);
    qkv_gemm_kernel<<<dim3((B_ * S_) / 16, D_ / 128), 256, 0, stream>>>(xb, WvT, bv, Vt, 1);

    // 3) fused scores + softmax + weights write + attn@V
    fused_attn_kernel<<<dim3(S_ / 16, B_ * H_), 256, 0, stream>>>(Qb, Kb, Vt, weights, attnb);

    // 4) output projection
    out_gemm_kernel<<<dim3((B_ * S_) / 16, D_ / 128), 256, 0, stream>>>(attnb, WoT, bo, out);
}
